// RelationNet_8375186227421
// MI455X (gfx1250) — matrix operand order repaired and checked by reading, not compiled, not run on hardware
//
#include <hip/hip_runtime.h>
#include <hip/hip_bf16.h>
#include <math.h>

typedef __attribute__((ext_vector_type(16))) __bf16 v16bf;
typedef __attribute__((ext_vector_type(8)))  __bf16 v8bf;
typedef __attribute__((ext_vector_type(8)))  float  v8f;

#define XSTRIDE 712   // bf16 elems; 712*2B -> lane m stride 36 banks, conflict-free for 16 lanes
#define H0STRIDE 264  // bf16 elems

__device__ __forceinline__ float gelu_exact(float x) {
    return 0.5f * x * (1.0f + erff(x * 0.70710678118654752440f));
}

// ---------------- K0: transpose f32 (K,N) -> bf16 [N][Kpad], zero-padded K ----------------
__global__ void transpose_to_bf16(const float* __restrict__ src, __bf16* __restrict__ dst,
                                  int K, int N, int Kpad) {
    int idx = blockIdx.x * blockDim.x + threadIdx.x;
    int total = N * Kpad;
    if (idx >= total) return;
    int n = idx / Kpad, k = idx - n * Kpad;
    float v = (k < K) ? src[k * N + n] : 0.0f;
    dst[n * Kpad + k] = (__bf16)v;
}

// ---------------- K1: gather + MLP0(gelu) + MLP1(gelu) via bf16 WMMA ----------------
// grid = 8192 (16 tuples each), block = 512 (16 waves)
__global__ __launch_bounds__(512) void mlp_kernel(
    const int* __restrict__ qa_ids, const int* __restrict__ rel_ids,
    const float* __restrict__ cemb, const float* __restrict__ remb,
    const __bf16* __restrict__ w0t,  // [256][704] bf16
    const float* __restrict__ b0,
    const __bf16* __restrict__ w1t,  // [128][256] bf16
    const float* __restrict__ b1,
    float* __restrict__ hout)        // [131072][128] f32
{
    __shared__ __bf16 X[16 * XSTRIDE];    // 16 x 704 (padded) concat features
    __shared__ __bf16 H0[16 * H0STRIDE];  // 16 x 256 hidden0
    __shared__ int ids[16 * 3];

    const int tid  = threadIdx.x;
    const int row0 = blockIdx.x * 16;

    if (tid < 16) {
        int row = row0 + tid;
        ids[tid * 3 + 0] = qa_ids[row * 2 + 0];
        ids[tid * 3 + 1] = qa_ids[row * 2 + 1];
        ids[tid * 3 + 2] = rel_ids[row];
    }
    __syncthreads();

    // Gather concat = [cemb[id0](300) | cemb[id1](300) | remb[rid](100) | 0-pad(4)]
    for (int i = tid; i < 16 * 704; i += 512) {
        int r = i / 704, c = i - r * 704;
        float v = 0.0f;
        if (c < 300)      v = cemb[(size_t)ids[r * 3 + 0] * 300 + c];
        else if (c < 600) v = cemb[(size_t)ids[r * 3 + 1] * 300 + (c - 300)];
        else if (c < 700) v = remb[(size_t)ids[r * 3 + 2] * 100 + (c - 600)];
        X[r * XSTRIDE + c] = (__bf16)v;
    }
    __syncthreads();

    const int wave = tid >> 5, lane = tid & 31;
    const int m = lane & 15, kh = lane >> 4;

    // ---- MLP0: X(16x704) @ W0(704x256), wave w -> output cols [16w,16w+16) ----
    {
        const int n0 = wave * 16;
        v8f acc = {};
        for (int kb = 0; kb < 704; kb += 32) {
            // A frag: K = 16kh + 0..15 of row m (contiguous, matches B)
            union { v16bf v; v8bf h[2]; } a;
            a.h[0] = *(const v8bf*)(X + m * XSTRIDE + kb + 16 * kh);
            a.h[1] = *(const v8bf*)(X + m * XSTRIDE + kb + 16 * kh + 8);
            // B frag: K = 16kh + 0..15 of column n0+m (contiguous in W0T)
            v16bf bfrag = *(const v16bf*)(w0t + (size_t)(n0 + m) * 704 + kb + 16 * kh);
            acc = __builtin_amdgcn_wmma_f32_16x16x32_bf16(
                false, a.v, false, bfrag, (short)0, acc, false, false);
        }
        // C layout: VGPR i -> row (i + 8*kh), col (n0 + m)
        float bias = b0[n0 + m];
        #pragma unroll
        for (int i = 0; i < 8; ++i) {
            int r = i + 8 * kh;
            H0[r * H0STRIDE + n0 + m] = (__bf16)gelu_exact(acc[i] + bias);
        }
    }
    __syncthreads();

    // ---- MLP1: H0(16x256) @ W1(256x128), waves 0..7 ----
    if (wave < 8) {
        const int n0 = wave * 16;
        v8f acc = {};
        for (int kb = 0; kb < 256; kb += 32) {
            union { v16bf v; v8bf h[2]; } a;
            a.h[0] = *(const v8bf*)(H0 + m * H0STRIDE + kb + 16 * kh);
            a.h[1] = *(const v8bf*)(H0 + m * H0STRIDE + kb + 16 * kh + 8);
            v16bf bfrag = *(const v16bf*)(w1t + (size_t)(n0 + m) * 256 + kb + 16 * kh);
            acc = __builtin_amdgcn_wmma_f32_16x16x32_bf16(
                false, a.v, false, bfrag, (short)0, acc, false, false);
        }
        float bias = b1[n0 + m];
        #pragma unroll
        for (int i = 0; i < 8; ++i) {
            int r = i + 8 * kh;
            hout[(size_t)(row0 + r) * 128 + n0 + m] = gelu_exact(acc[i] + bias);
        }
    }
}

// ---------------- K2: per-batch attention pooling + logits (f32, reassociated) ----------------
// grid = 256 (one per batch row), block = 256 (8 waves)
__global__ __launch_bounds__(256) void attn_kernel(
    const float* __restrict__ sent, const float* __restrict__ h,
    const int* __restrict__ num_tuples,
    const float* __restrict__ wq, const float* __restrict__ bq,
    const float* __restrict__ wk, const float* __restrict__ bk,
    const float* __restrict__ wv, const float* __restrict__ bv,
    const float* __restrict__ fcw, const float* __restrict__ fcb,
    float* __restrict__ out)  // [0..255] logits, [256..] attn (head*256+b)*512+l
{
    __shared__ float qs[128];
    __shared__ float qk[2][128];
    __shared__ float sbias[2];
    __shared__ float sc[2][512];
    __shared__ float red[256];
    __shared__ float hw[2][128];
    __shared__ float pooled[128];

    const int b = blockIdx.x, tid = threadIdx.x;
    const float* hb = h + (size_t)b * 512 * 128;
    const float* sb = sent + (size_t)b * 1024;

    // q = sent @ Wq + bq   (128 outputs)
    if (tid < 128) {
        float acc = bq[tid];
        for (int k = 0; k < 1024; ++k) acc += sb[k] * wq[k * 128 + tid];
        qs[tid] = acc;
    }
    __syncthreads();

    // Fold Wk into q: qk[hd][c] = sum_d q[hd,d] * Wk[c, hd*64+d]
    {
        int hd = tid >> 7, c = tid & 127;
        float acc = 0.0f;
        const float* qh = &qs[hd * 64];
        for (int d = 0; d < 64; ++d) acc += qh[d] * wk[c * 128 + hd * 64 + d];
        qk[hd][c] = acc;
    }
    if (tid < 2) {
        float acc = 0.0f;
        for (int d = 0; d < 64; ++d) acc += qs[tid * 64 + d] * bk[tid * 64 + d];
        sbias[tid] = acc;
    }
    __syncthreads();

    int nt = num_tuples[b]; if (nt < 1) nt = 1;

    // scores[hd][l] = (h[l,:] . qk[hd] + q.bk) / sqrt(64), masked
    const int wave = tid >> 5, lane = tid & 31;
    for (int l = wave; l < 512; l += 8) {
        float p0 = 0.0f, p1 = 0.0f;
        const float* hr = hb + (size_t)l * 128;
        for (int j = lane; j < 128; j += 32) {
            float hv = hr[j];
            p0 += hv * qk[0][j];
            p1 += hv * qk[1][j];
        }
        for (int off = 16; off; off >>= 1) {
            p0 += __shfl_down(p0, off, 32);
            p1 += __shfl_down(p1, off, 32);
        }
        if (lane == 0) {
            bool pad = (l >= nt);
            sc[0][l] = pad ? -INFINITY : (p0 + sbias[0]) * 0.125f;
            sc[1][l] = pad ? -INFINITY : (p1 + sbias[1]) * 0.125f;
        }
    }
    __syncthreads();

    // masked softmax per head; write att_scores
    for (int hd = 0; hd < 2; ++hd) {
        float lm = -INFINITY;
        for (int l = tid; l < 512; l += 256) lm = fmaxf(lm, sc[hd][l]);
        red[tid] = lm; __syncthreads();
        for (int s = 128; s; s >>= 1) { if (tid < s) red[tid] = fmaxf(red[tid], red[tid + s]); __syncthreads(); }
        float mx = red[0]; __syncthreads();

        float ls = 0.0f;
        for (int l = tid; l < 512; l += 256) { float e = expf(sc[hd][l] - mx); sc[hd][l] = e; ls += e; }
        red[tid] = ls; __syncthreads();
        for (int s = 128; s; s >>= 1) { if (tid < s) red[tid] += red[tid + s]; __syncthreads(); }
        float inv = 1.0f / red[0]; __syncthreads();

        for (int l = tid; l < 512; l += 256) {
            float a = sc[hd][l] * inv;
            sc[hd][l] = a;
            out[256 + ((size_t)hd * 256 + b) * 512 + l] = a;
        }
        __syncthreads();
    }

    // hw[hd][c] = sum_l attn[hd][l] * h[l][c]
    {
        int hd = tid >> 7, c = tid & 127;
        float acc = 0.0f;
        for (int l = 0; l < 512; ++l) acc += sc[hd][l] * hb[(size_t)l * 128 + c];
        hw[hd][c] = acc;
    }
    __syncthreads();

    // pooled[j] = hw[j/64][:] . Wv[:, j] + bv[j]
    if (tid < 128) {
        int hd = tid >> 6;
        float acc = bv[tid];
        for (int c = 0; c < 128; ++c) acc += hw[hd][c] * wv[c * 128 + tid];
        pooled[tid] = acc;
    }
    __syncthreads();

    // logits[b] = [pooled | sent] . fc_w + fc_b
    float part = 0.0f;
    for (int i = tid; i < 1152; i += 256) {
        float v = (i < 128) ? pooled[i] : sb[i - 128];
        part += v * fcw[i];
    }
    red[tid] = part; __syncthreads();
    for (int s = 128; s; s >>= 1) { if (tid < s) red[tid] += red[tid + s]; __syncthreads(); }
    if (tid == 0) out[b] = red[0] + fcb[0];
}

extern "C" void kernel_launch(void* const* d_in, const int* in_sizes, int n_in,
                              void* d_out, int out_size, void* d_ws, size_t ws_size,
                              hipStream_t stream) {
    const float* sent = (const float*)d_in[0];
    const int*   qa   = (const int*)d_in[1];
    const int*   rel  = (const int*)d_in[2];
    const int*   ntup = (const int*)d_in[3];
    const float* cemb = (const float*)d_in[4];
    const float* remb = (const float*)d_in[5];
    const float* w0   = (const float*)d_in[6];
    const float* b0   = (const float*)d_in[7];
    const float* w1   = (const float*)d_in[8];
    const float* b1   = (const float*)d_in[9];
    const float* wq   = (const float*)d_in[10];
    const float* bq   = (const float*)d_in[11];
    const float* wk   = (const float*)d_in[12];
    const float* bk   = (const float*)d_in[13];
    const float* wv   = (const float*)d_in[14];
    const float* bv   = (const float*)d_in[15];
    const float* fcw  = (const float*)d_in[16];
    const float* fcb  = (const float*)d_in[17];
    float* out = (float*)d_out;

    char* ws = (char*)d_ws;
    float*  h   = (float*)ws;                                   // 131072*128*4 = 64 MB
    __bf16* w0t = (__bf16*)(ws + (size_t)67108864);             // 256*704*2
    __bf16* w1t = (__bf16*)(ws + (size_t)67108864 + 360448);    // 128*256*2

    transpose_to_bf16<<<(256 * 704 + 255) / 256, 256, 0, stream>>>(w0, w0t, 700, 256, 704);
    transpose_to_bf16<<<(128 * 256 + 255) / 256, 256, 0, stream>>>(w1, w1t, 256, 128, 256);
    mlp_kernel<<<8192, 512, 0, stream>>>(qa, rel, cemb, remb, w0t, b0, w1t, b1, h);
    attn_kernel<<<256, 256, 0, stream>>>(sent, h, ntup, wq, bq, wk, bk, wv, bv, fcw, fcb, out);
}
